// CannyEdgeLoss_3341484557054
// MI455X (gfx1250) — compile-verified, matrix operand order checked by reading, not run on hardware
//
#include <hip/hip_runtime.h>
#include <hip/hip_bf16.h>
#include <stdint.h>

// ---------------- CDNA5 feature detection ----------------
#if __has_builtin(__builtin_amdgcn_global_load_async_to_lds_b32)
#define HAVE_ASYNC_LDS 1
#else
#define HAVE_ASYNC_LDS 0
#endif

#define AS_GLOBAL __attribute__((address_space(1)))
#define AS_LDS    __attribute__((address_space(3)))

__device__ __forceinline__ void async_wait_all() {
#if HAVE_ASYNC_LDS
#if __has_builtin(__builtin_amdgcn_s_wait_asynccnt)
  __builtin_amdgcn_s_wait_asynccnt(0);
#else
  asm volatile("s_wait_asynccnt 0" ::: "memory");
#endif
#endif
}

__device__ __forceinline__ void async_load_f32(const float* g, float* lds) {
#if HAVE_ASYNC_LDS
  __builtin_amdgcn_global_load_async_to_lds_b32(
      (AS_GLOBAL int*)(float*)g, (AS_LDS int*)lds, 0, 0);
#else
  *lds = *g;
#endif
}

__device__ __forceinline__ void async_load_u8(const unsigned char* g, unsigned char* lds) {
#if HAVE_ASYNC_LDS && __has_builtin(__builtin_amdgcn_global_load_async_to_lds_b8)
  __builtin_amdgcn_global_load_async_to_lds_b8(
      (AS_GLOBAL char*)(unsigned char*)g, (AS_LDS char*)lds, 0, 0);
#else
  *lds = *g;
#endif
}

// ---------------- problem constants ----------------
#define BB 16
#define HH 512
#define WW 512
#define HW (HH * WW)
#define NTOT (BB * HW)
#define TILE 32

__device__ __forceinline__ int reflectI(int i, int n) {
  if (i < 0) i = -i;
  if (i >= n) i = 2 * n - 2 - i;
  return i;
}
__device__ __forceinline__ int clampI(int i, int n) {
  return i < 0 ? 0 : (i >= n ? n - 1 : i);
}

// ---------------- K1: gray = mean_c(pred*matte) ----------------
__global__ __launch_bounds__(256) void k_gray(const float* __restrict__ pred,
                                              const float* __restrict__ matte,
                                              float* __restrict__ gray) {
  size_t i = (size_t)blockIdx.x * 256 + threadIdx.x;
  if (i >= (size_t)NTOT) return;
  size_t b = i >> 18;            // / HW (2^18)
  size_t p = i & (HW - 1);
  const float* pr = pred + b * (3 * (size_t)HW) + p;
  float m = matte[i];
  gray[i] = (pr[0] + pr[HW] + pr[2 * (size_t)HW]) * m * (1.0f / 3.0f);
}

// ---------------- K2: fused blur(reflect) + sobel(edge) + NMS(zero) + threshold --------
// gaussian 5x5 sigma=1 separable weights
#define GW0 0.05448868454f
#define GW1 0.24420134200f
#define GW2 0.40261994692f

__global__ __launch_bounds__(256) void k_edges(const float* __restrict__ gray,
                                               unsigned char* __restrict__ edges) {
  __shared__ float sG[40][40];   // gray, reflect-padded window  rows ty0-4..ty0+35
  __shared__ float sT[40][36];   // horizontal blur temp
  __shared__ float sB[36][36];   // blurred, replicate-extended   rows ty0-2..ty0+33
  __shared__ float sM[34][34];   // magnitude (0 outside image)   rows ty0-1..ty0+32

  const int tx0 = blockIdx.x * TILE, ty0 = blockIdx.y * TILE, b = blockIdx.z;
  const int gy0 = ty0 - 4, gx0 = tx0 - 4;
  const int tid = threadIdx.x;
  const float* g = gray + (size_t)b * HW;

  // stage gray window (reflect padding of the original image) via async-to-LDS
  for (int idx = tid; idx < 40 * 40; idx += 256) {
    int i = idx / 40, j = idx - i * 40;
    int r = reflectI(gy0 + i, HH);
    int c = reflectI(gx0 + j, WW);
    async_load_f32(g + (size_t)r * WW + c, &sG[i][j]);
  }
  async_wait_all();
  __syncthreads();

  // horizontal gaussian pass (columns already clamped for replicate-extension of blurred)
  for (int idx = tid; idx < 40 * 36; idx += 256) {
    int i = idx / 36, j = idx - i * 36;
    int jc = clampI(tx0 - 2 + j, WW) - gx0;   // in [2,37]
    sT[i][j] = GW0 * (sG[i][jc - 2] + sG[i][jc + 2]) +
               GW1 * (sG[i][jc - 1] + sG[i][jc + 1]) +
               GW2 * sG[i][jc];
  }
  __syncthreads();

  // vertical gaussian pass -> sB[i][j] = blurred[clamp(ty0-2+i)][clamp(tx0-2+j)]
  for (int idx = tid; idx < 36 * 36; idx += 256) {
    int i = idx / 36, j = idx - i * 36;
    int ic = clampI(ty0 - 2 + i, HH) - gy0;   // in [2,37]
    sB[i][j] = GW0 * (sT[ic - 2][j] + sT[ic + 2][j]) +
               GW1 * (sT[ic - 1][j] + sT[ic + 1][j]) +
               GW2 * sT[ic][j];
  }
  __syncthreads();

  // sobel magnitude on tile+1 halo; zero outside image (NMS zero padding)
  for (int idx = tid; idx < 34 * 34; idx += 256) {
    int i = idx / 34, j = idx - i * 34;
    int r = ty0 - 1 + i, c = tx0 - 1 + j;
    float m = 0.0f;
    if (r >= 0 && r < HH && c >= 0 && c < WW) {
      float a00 = sB[i][j],     a01 = sB[i][j + 1],     a02 = sB[i][j + 2];
      float a10 = sB[i + 1][j],                         a12 = sB[i + 1][j + 2];
      float a20 = sB[i + 2][j], a21 = sB[i + 2][j + 1], a22 = sB[i + 2][j + 2];
      float gx = (a02 - a00 + 2.0f * (a12 - a10) + a22 - a20) * 0.125f;
      float gy = (a20 - a00 + 2.0f * (a21 - a01) + a22 - a02) * 0.125f;
      m = sqrtf(gx * gx + gy * gy + 1e-6f);
    }
    sM[i][j] = m;
  }
  __syncthreads();

  // per-output-pixel: direction, NMS, double threshold -> u8 code {0,1,2}
  const int lx = tid & 31, tyb = tid >> 5;
  const int di[8] = {0, -1, -1, -1, 0, 1, 1, 1};
  const int dj[8] = {1, 1, 0, -1, -1, -1, 0, 1};
#pragma unroll
  for (int k = 0; k < 4; ++k) {
    int ly = tyb + 8 * k;
    int r = ty0 + ly, c = tx0 + lx;
    float a00 = sB[ly + 1][lx + 1], a01 = sB[ly + 1][lx + 2], a02 = sB[ly + 1][lx + 3];
    float a10 = sB[ly + 2][lx + 1],                           a12 = sB[ly + 2][lx + 3];
    float a20 = sB[ly + 3][lx + 1], a21 = sB[ly + 3][lx + 2], a22 = sB[ly + 3][lx + 3];
    float gx = (a02 - a00 + 2.0f * (a12 - a10) + a22 - a20) * 0.125f;
    float gy = (a20 - a00 + 2.0f * (a21 - a01) + a22 - a02) * 0.125f;
    float mag = sM[ly + 1][lx + 1];
    float deg = atan2f(gy, gx) * 57.29577951308232f;
    int kk = (int)rintf(deg * (1.0f / 45.0f));   // [-4,4], round half-to-even like jnp.round
    int pos = kk & 7;
    int neg = (pos + 4) & 7;
    float np = sM[ly + 1 + di[pos]][lx + 1 + dj[pos]];
    float nn = sM[ly + 1 + di[neg]][lx + 1 + dj[neg]];
    float cp = mag - np, cn = mag - nn;
    float m2 = (fminf(cp, cn) > 0.0f) ? mag : 0.0f;
    unsigned char code = (unsigned char)((m2 > 0.1f) + (m2 > 0.2f));
    edges[(size_t)b * HW + (size_t)r * WW + c] = code;
  }
}

// ---------------- K3: hysteresis sweep (intra-tile fixpoint in LDS) ----------------
__global__ __launch_bounds__(256) void k_hyst(unsigned char* edges) {
  __shared__ unsigned char s[34][34];
  __shared__ int changed;
  const int tx0 = blockIdx.x * TILE, ty0 = blockIdx.y * TILE, b = blockIdx.z;
  const int tid = threadIdx.x;
  unsigned char* e = edges + (size_t)b * HW;

  for (int idx = tid; idx < 34 * 34; idx += 256) {
    int i = idx / 34, j = idx - i * 34;
    int r = ty0 - 1 + i, c = tx0 - 1 + j;
    if (r >= 0 && r < HH && c >= 0 && c < WW) {
      async_load_u8(e + (size_t)r * WW + c, &s[i][j]);
    } else {
      s[i][j] = 0;
    }
  }
  async_wait_all();
  __syncthreads();

  const int lx = tid & 31, tyb = tid >> 5;
  for (int it = 0; it < 64; ++it) {
    if (tid == 0) changed = 0;
    __syncthreads();
    bool any = false;
#pragma unroll
    for (int k = 0; k < 4; ++k) {
      int i = tyb + 8 * k + 1, j = lx + 1;
      if (s[i][j] == 1) {
        bool strong = s[i - 1][j - 1] == 2 || s[i - 1][j] == 2 || s[i - 1][j + 1] == 2 ||
                      s[i][j - 1] == 2     ||                     s[i][j + 1] == 2 ||
                      s[i + 1][j - 1] == 2 || s[i + 1][j] == 2 || s[i + 1][j + 1] == 2;
        if (strong) { s[i][j] = 2; any = true; }
      }
    }
    if (any) changed = 1;
    __syncthreads();
    if (!changed) break;
  }

#pragma unroll
  for (int k = 0; k < 4; ++k) {
    int ly = tyb + 8 * k;
    e[(size_t)(ty0 + ly) * WW + (tx0 + lx)] = s[ly + 1][lx + 1];
  }
}

// ---------------- K4: MSE partials, wave reduced through WMMA ----------------
typedef __attribute__((ext_vector_type(16))) _Float16 v16h;
typedef __attribute__((ext_vector_type(8)))  float    v8f;

#define RED_BLOCKS 1024

__global__ __launch_bounds__(256) void k_loss(const unsigned char* __restrict__ edges,
                                              const float* __restrict__ sketch,
                                              const float* __restrict__ matte,
                                              float* __restrict__ partials) {
  float acc = 0.0f;
  for (size_t i = (size_t)blockIdx.x * 256 + threadIdx.x; i < (size_t)NTOT;
       i += (size_t)RED_BLOCKS * 256) {
    float t = sketch[i] * matte[i];
    t = fminf(fmaxf(t, 0.0f), 1.0f);           // resize(identity) + clip
    float ev = 0.5f * (float)edges[i];
    float d = ev - t;
    acc += d * d;
  }

  // Fold the wave's 32 partials with V_WMMA_F32_16X16X32_F16:
  // A[lane-row][K=0] = acc (f16), B = ones -> D row-sums; f32 accumulate.
  v16h va;
#pragma unroll
  for (int i = 0; i < 16; ++i) va[i] = (_Float16)0.0f;
  va[0] = (_Float16)acc;
  v16h vb;
#pragma unroll
  for (int i = 0; i < 16; ++i) vb[i] = (_Float16)1.0f;
  v8f vc = {0.f, 0.f, 0.f, 0.f, 0.f, 0.f, 0.f, 0.f};
  vc = __builtin_amdgcn_wmma_f32_16x16x32_f16(false, va, false, vb,
                                              (short)0, vc, false, false);
  float csum = vc[0] + vc[1] + vc[2] + vc[3] + vc[4] + vc[5] + vc[6] + vc[7];
  csum += __shfl_xor(csum, 16, 32);            // rows 0-7 half + rows 8-15 half

  __shared__ float wpart[8];
  int wid = threadIdx.x >> 5;
  if ((threadIdx.x & 31) == 0) wpart[wid] = csum;
  __syncthreads();
  if (threadIdx.x == 0) {
    float t = 0.0f;
#pragma unroll
    for (int i = 0; i < 8; ++i) t += wpart[i];
    partials[blockIdx.x] = t;
  }
}

__global__ __launch_bounds__(256) void k_final(const float* __restrict__ partials,
                                               float* __restrict__ out) {
  __shared__ float sh[256];
  float a = 0.0f;
  for (int i = threadIdx.x; i < RED_BLOCKS; i += 256) a += partials[i];
  sh[threadIdx.x] = a;
  __syncthreads();
  for (int s = 128; s > 0; s >>= 1) {
    if (threadIdx.x < (unsigned)s) sh[threadIdx.x] += sh[threadIdx.x + s];
    __syncthreads();
  }
  if (threadIdx.x == 0) out[0] = sh[0] * (1.0f / (float)NTOT);
}

// ---------------- launch ----------------
extern "C" void kernel_launch(void* const* d_in, const int* in_sizes, int n_in,
                              void* d_out, int out_size, void* d_ws, size_t ws_size,
                              hipStream_t stream) {
  const float* pred   = (const float*)d_in[0];
  const float* sketch = (const float*)d_in[1];
  const float* matte  = (const float*)d_in[2];
  float* out = (float*)d_out;

  // workspace layout: gray f32[N] | edges u8[N] | partials f32[1024]  (~21 MB)
  float* gray = (float*)d_ws;
  unsigned char* edges = (unsigned char*)((char*)d_ws + (size_t)NTOT * 4);
  float* partials = (float*)((char*)d_ws + (size_t)NTOT * 4 + (size_t)NTOT);

  k_gray<<<(NTOT + 255) / 256, 256, 0, stream>>>(pred, matte, gray);

  dim3 tg(WW / TILE, HH / TILE, BB);
  k_edges<<<tg, 256, 0, stream>>>(gray, edges);

  for (int i = 0; i < 48; ++i)
    k_hyst<<<tg, 256, 0, stream>>>(edges);

  k_loss<<<RED_BLOCKS, 256, 0, stream>>>(edges, sketch, matte, partials);
  k_final<<<1, 256, 0, stream>>>(partials, out);
}
